// ColBERT_8976481649168
// MI455X (gfx1250) — matrix operand order repaired and checked by reading, not compiled, not run on hardware
//
#include <hip/hip_runtime.h>
#include <hip/hip_bf16.h>

// ColBERT late-interaction pipeline for MI455X (gfx1250, wave32, WMMA).
// Shapes: B=64, LQ=32, LP=256, H=768, D=128.

typedef __attribute__((ext_vector_type(16))) __bf16 v16bf;
typedef __attribute__((ext_vector_type(8)))  __bf16 v8bf;   // 16 bytes
typedef __attribute__((ext_vector_type(8)))  float  v8f;

#define B_   64
#define LQ_  32
#define LP_  256
#define H_   768
#define D_   128
#define NEGHUGE (-3.4e38f)

union bfu { v16bf v; v8bf h[2]; };

// Two aligned 16-byte loads -> one 16-element bf16 fragment (no element movs).
__device__ __forceinline__ v16bf ld_frag(const __bf16* p0, const __bf16* p1) {
  bfu u;
  u.h[0] = *(const v8bf*)p0;
  u.h[1] = *(const v8bf*)p1;
  return u.v;
}

// ---------------------------------------------------------------------------
// Stage 1: projection GEMM  X[M,128] = H[M,768] @ W[768,128] + bias
// bf16 WMMA 16x16x32, f32 accumulate. One block (8 waves) = 128 rows x 128 cols.
// W K-slice staged in LDS *transposed* [n][k] with padded stride so each lane's
// B-fragment is two contiguous, conflict-free ds_load_b128.
// ---------------------------------------------------------------------------
#define WT_STRIDE 40   // bf16 elems per LDS row: 80 B, 16 B aligned, 20 dwords
__global__ __launch_bounds__(256) void project_kernel(
    const float* __restrict__ h, const float* __restrict__ W,
    const float* __restrict__ bias, float* __restrict__ xout) {
  __shared__ __bf16 Wt[D_ * WT_STRIDE];       // 128 x 40 bf16 = 10 KB
  const int lane = threadIdx.x & 31;
  const int wave = threadIdx.x >> 5;
  const int half = lane >> 4;                 // 0: lanes 0-15, 1: lanes 16-31
  const int r    = lane & 15;
  const long rowbase = (long)blockIdx.x * 128 + wave * 16;
  const float* arow = h + (rowbase + r) * H_; // A row for this lane

  v8f zero = {};
  v8f acc[8];
#pragma unroll
  for (int j = 0; j < 8; ++j) acc[j] = zero;

  for (int k0 = 0; k0 < H_; k0 += 32) {
    // Cooperative transposed load: Wt[n][k] = W[k0+k][n] as bf16.
    // Global side coalesced over n; LDS side sub-dword writes, padded stride.
    for (int t = threadIdx.x; t < 32 * D_; t += 256) {
      const int k = t >> 7, n = t & 127;
      Wt[n * WT_STRIDE + k] = (__bf16)W[(size_t)(k0 + k) * D_ + n];
    }
    __syncthreads();

    // A-frag (16-bit A 16x32 layout): lanes 0-15 hold K 0..7,16..23;
    // lanes 16-31 hold K 8..15,24..31.  Two 32-B f32 chunks, converted.
    const float4 f0 = *(const float4*)(arow + k0 + 8 * half);
    const float4 f1 = *(const float4*)(arow + k0 + 8 * half + 4);
    const float4 f2 = *(const float4*)(arow + k0 + 8 * half + 16);
    const float4 f3 = *(const float4*)(arow + k0 + 8 * half + 20);
    v16bf a;
    a[0]  = (__bf16)f0.x; a[1]  = (__bf16)f0.y; a[2]  = (__bf16)f0.z; a[3]  = (__bf16)f0.w;
    a[4]  = (__bf16)f1.x; a[5]  = (__bf16)f1.y; a[6]  = (__bf16)f1.z; a[7]  = (__bf16)f1.w;
    a[8]  = (__bf16)f2.x; a[9]  = (__bf16)f2.y; a[10] = (__bf16)f2.z; a[11] = (__bf16)f2.w;
    a[12] = (__bf16)f3.x; a[13] = (__bf16)f3.y; a[14] = (__bf16)f3.z; a[15] = (__bf16)f3.w;

#pragma unroll
    for (int j = 0; j < 8; ++j) {
      // B-frag: b[i] = Wt[n][8*half + i] for i<8, Wt[n][16 + 8*half + i-8] for i>=8 (matches A)
      const __bf16* wrow = &Wt[(j * 16 + r) * WT_STRIDE + 8 * half];
      const v16bf bf = ld_frag(wrow, wrow + 16);
      acc[j] = __builtin_amdgcn_wmma_f32_16x16x32_bf16(
          false, a, false, bf, (short)0, acc[j], false, false);
    }
    __syncthreads();
  }

  // C layout: lane (half,r), vgpr v holds C[m = v + 8*half][n = r]
#pragma unroll
  for (int j = 0; j < 8; ++j) {
#pragma unroll
    for (int v = 0; v < 8; ++v) {
      const int col = j * 16 + r;
      const long row = rowbase + v + 8 * half;
      xout[row * D_ + col] = acc[j][v] + bias[col];
    }
  }
}

// ---------------------------------------------------------------------------
// Stage 2: per-(batch, d) L2 norm over the TOKEN axis (faithful to dim=1),
// scale, and convert to bf16 for the interaction stage.
// ---------------------------------------------------------------------------
__global__ __launch_bounds__(128) void normalize_kernel(
    const float* __restrict__ x, __bf16* __restrict__ out, int L) {
  const int bb = blockIdx.x;
  const int d  = threadIdx.x;
  const float* xb = x + (size_t)bb * L * D_;
  float s = 0.f;
  for (int l = 0; l < L; ++l) {
    float v = xb[(size_t)l * D_ + d];
    s += v * v;
  }
  const float inv = 1.0f / fmaxf(sqrtf(s), 1e-12f);
  __bf16* ob = out + (size_t)bb * L * D_;
  for (int l = 0; l < L; ++l)
    ob[(size_t)l * D_ + d] = (__bf16)(xb[(size_t)l * D_ + d] * inv);
}

// ---------------------------------------------------------------------------
// Stage 3: MaxSim. One wave per (query batch b, passage batch c) pair:
// sims[32,256] = q_b[32,128] @ p_c[256,128]^T via bf16 WMMA, masked max over
// passage tokens, cross-lane max reduction, sum over 32 query tokens.
// All fragments loaded as aligned 16-byte vectors (global_load_b128).
// ---------------------------------------------------------------------------
__global__ __launch_bounds__(256) void maxsim_kernel(
    const __bf16* __restrict__ qbf, const __bf16* __restrict__ pbf,
    const int* __restrict__ mask, float* __restrict__ out, int colOff) {
  const int wid  = blockIdx.x * 8 + (threadIdx.x >> 5);
  const int b    = wid >> 6;
  const int c    = wid & 63;
  const int lane = threadIdx.x & 31;
  const int half = lane >> 4;
  const int r    = lane & 15;

  const __bf16* q  = qbf + (size_t)b * LQ_ * D_;
  const __bf16* p  = pbf + (size_t)c * LP_ * D_;
  const int*    mk = mask + c * LP_;

  // Preload all A fragments: 2 M-tiles (32 q tokens) x 4 K-steps (128 dims)
  v16bf afr[2][4];
#pragma unroll
  for (int mt = 0; mt < 2; ++mt) {
    const __bf16* qrow = q + (size_t)(mt * 16 + r) * D_;
#pragma unroll
    for (int k = 0; k < 4; ++k)
      afr[mt][k] = ld_frag(qrow + k * 32 + 8 * half,
                           qrow + k * 32 + 8 * half + 16);
  }

  float vmax[2][8];
#pragma unroll
  for (int mt = 0; mt < 2; ++mt)
#pragma unroll
    for (int v = 0; v < 8; ++v) vmax[mt][v] = NEGHUGE;

  v8f zero = {};
  for (int nt = 0; nt < 16; ++nt) {          // 16 N-tiles cover 256 passage tokens
    const __bf16* prow = p + (size_t)(nt * 16 + r) * D_; // B[k][n] = p[n][k]
    v8f acc0 = zero, acc1 = zero;
#pragma unroll
    for (int k = 0; k < 4; ++k) {
      const v16bf bf = ld_frag(prow + k * 32 + 8 * half,
                               prow + k * 32 + 8 * half + 16);
      acc0 = __builtin_amdgcn_wmma_f32_16x16x32_bf16(
          false, afr[0][k], false, bf, (short)0, acc0, false, false);
      acc1 = __builtin_amdgcn_wmma_f32_16x16x32_bf16(
          false, afr[1][k], false, bf, (short)0, acc1, false, false);
    }
    // mask over passage token n = nt*16 + r (same column for both halves)
    const bool valid = mk[nt * 16 + r] != 0;
#pragma unroll
    for (int v = 0; v < 8; ++v) {
      vmax[0][v] = fmaxf(vmax[0][v], valid ? acc0[v] : NEGHUGE);
      vmax[1][v] = fmaxf(vmax[1][v], valid ? acc1[v] : NEGHUGE);
    }
  }

  // Max over the 16 column-lanes inside each half-group (rows live in (mt,half,v)),
  // then sum over the 16 rows this half holds, then add the other half's partial.
  float partial = 0.f;
#pragma unroll
  for (int mt = 0; mt < 2; ++mt)
#pragma unroll
    for (int v = 0; v < 8; ++v) {
      float x = vmax[mt][v];
#pragma unroll
      for (int mm = 1; mm < 16; mm <<= 1)
        x = fmaxf(x, __shfl_xor(x, mm, 32));
      partial += x;
    }
  const float score = partial + __shfl_xor(partial, 16, 32);
  if (lane == 0) out[b * (2 * B_) + colOff + c] = score;
}

// ---------------------------------------------------------------------------
extern "C" void kernel_launch(void* const* d_in, const int* in_sizes, int n_in,
                              void* d_out, int out_size, void* d_ws, size_t ws_size,
                              hipStream_t stream) {
  const float* qh    = (const float*)d_in[0];
  const float* ph    = (const float*)d_in[1];
  const float* nh    = (const float*)d_in[2];
  const float* W     = (const float*)d_in[3];
  const float* bias  = (const float*)d_in[4];
  const int*   pmask = (const int*)d_in[5];
  const int*   nmask = (const int*)d_in[6];
  float* out = (float*)d_out;

  // Workspace layout: f32 projections then bf16 normalized vectors (~25.5 MB)
  float* xq = (float*)d_ws;
  float* xp = xq + (size_t)B_ * LQ_ * D_;
  float* xn = xp + (size_t)B_ * LP_ * D_;
  __bf16* qbf = (__bf16*)(xn + (size_t)B_ * LP_ * D_);
  __bf16* pbf = qbf + (size_t)B_ * LQ_ * D_;
  __bf16* nbf = pbf + (size_t)B_ * LP_ * D_;

  project_kernel<<<(B_ * LQ_) / 128, 256, 0, stream>>>(qh, W, bias, xq);
  project_kernel<<<(B_ * LP_) / 128, 256, 0, stream>>>(ph, W, bias, xp);
  project_kernel<<<(B_ * LP_) / 128, 256, 0, stream>>>(nh, W, bias, xn);

  normalize_kernel<<<B_, D_, 0, stream>>>(xq, qbf, LQ_);
  normalize_kernel<<<B_, D_, 0, stream>>>(xp, pbf, LP_);
  normalize_kernel<<<B_, D_, 0, stream>>>(xn, nbf, LP_);

  maxsim_kernel<<<(B_ * B_) / 8, 256, 0, stream>>>(qbf, pbf, pmask, out, 0);
  maxsim_kernel<<<(B_ * B_) / 8, 256, 0, stream>>>(qbf, nbf, nmask, out, B_);
}
